// Seq2SeqNet_73100343378137
// MI455X (gfx1250) — matrix operand order repaired and checked by reading, not compiled, not run on hardware
//
#include <hip/hip_runtime.h>
#include <hip/hip_bf16.h>

// ---------------------------------------------------------------------------
// Seq2Seq GRU (S=T=2048, H=2048, V=256) for MI455X / gfx1250 (wave32, WMMA).
//
// Phases:
//   0. memset: grid-barrier counters + h ping-pong = 0
//   1. prep_convert:  X_bf16[S,H] = bf16(emb[inputs]),  enc_w_ih / enc_w_hh /
//                     dec_w_hh / fc_w -> bf16
//   2. gemm_bf16_wmma: GI_enc[S,3H] = X @ enc_w_ih^T + b_ih   (v_wmma bf16)
//   3. gi_dec_kernel:  gi_dec0 = dec_w_ih@emb[0]+b,  gi_decc = dec_w_ih@emb[targets[1]]+b
//   4. gru_recurrence: persistent kernel, 4096 sequential steps with a
//                      grid-wide barrier per step; whh streamed from L2 in
//                      bf16; h staged into LDS via async-to-LDS DMA
//   5. gemm_bf16_wmma: out[T,V] = dec_hs @ fc_w^T + fc_b       (v_wmma bf16)
// ---------------------------------------------------------------------------

#define Hh   2048
#define H3   6144
#define SS   2048
#define TT   2048
#define VV   256
#define NBLK 64          // persistent-kernel workgroups (co-resident, safe)

typedef __attribute__((ext_vector_type(16))) __bf16 v16bf;
typedef __attribute__((ext_vector_type(8)))  float  v8f;
typedef __attribute__((ext_vector_type(4)))  int    v4i;

struct U4  { unsigned a, b, c, d; };
struct R32 { unsigned u[8]; };

#if __has_builtin(__builtin_amdgcn_global_load_async_to_lds_b128)
#define HAVE_ASYNC_LDS 1
// builtin signature (from clang diagnostic): (v4i addrspace(1)*, v4i addrspace(3)*, imm offset, imm cpol)
#define AS1V4(p) ((__attribute__((address_space(1))) v4i*)(p))
#define AS3V4(p) ((__attribute__((address_space(3))) v4i*)(p))
#endif

static __device__ inline unsigned short f2bf(float f) {   // RNE fp32 -> bf16
  union { float f; unsigned u; } v; v.f = f;
  unsigned r = v.u + 0x7FFFu + ((v.u >> 16) & 1u);
  return (unsigned short)(r >> 16);
}
static __device__ inline float bflo(unsigned u) {
  union { unsigned v; float f; } t; t.v = u << 16;        return t.f;
}
static __device__ inline float bfhi(unsigned u) {
  union { unsigned v; float f; } t; t.v = u & 0xffff0000u; return t.f;
}

static __device__ inline v16bf make_frag(U4 lo, U4 hi) {
  R32 r;
  r.u[0] = lo.a; r.u[1] = lo.b; r.u[2] = lo.c; r.u[3] = lo.d;
  r.u[4] = hi.a; r.u[5] = hi.b; r.u[6] = hi.c; r.u[7] = hi.d;
  return __builtin_bit_cast(v16bf, r);
}

// ---------------------------------------------------------------------------
// 1. gather + fp32->bf16 conversion (flat index over 5 regions)
// ---------------------------------------------------------------------------
__global__ __launch_bounds__(256) void prep_convert(
    const int*   __restrict__ inputs,
    const float* __restrict__ emb,
    const float* __restrict__ enc_w_ih,
    const float* __restrict__ enc_w_hh,
    const float* __restrict__ dec_w_hh,
    const float* __restrict__ fc_w,
    unsigned short* __restrict__ xbf,
    unsigned short* __restrict__ wihbf,
    unsigned short* __restrict__ whhe,
    unsigned short* __restrict__ whhd,
    unsigned short* __restrict__ fcbf)
{
  const size_t i = (size_t)blockIdx.x * 256u + threadIdx.x;
  const size_t R0 = (size_t)SS * Hh;          //  4,194,304  X gather
  const size_t RW = (size_t)H3 * Hh;          // 12,582,912  per weight
  const size_t R1 = R0 + RW, R2 = R1 + RW, R3 = R2 + RW;
  const size_t R4 = R3 + (size_t)VV * Hh;
  if (i < R0) {
    const size_t s = i >> 11, c = i & (Hh - 1);
    xbf[i] = f2bf(emb[(size_t)inputs[s] * Hh + c]);
  } else if (i < R1) { const size_t j = i - R0; wihbf[j] = f2bf(enc_w_ih[j]); }
    else if (i < R2) { const size_t j = i - R1; whhe[j]  = f2bf(enc_w_hh[j]); }
    else if (i < R3) { const size_t j = i - R2; whhd[j]  = f2bf(dec_w_hh[j]); }
    else if (i < R4) { const size_t j = i - R3; fcbf[j]  = f2bf(fc_w[j]); }
}

// ---------------------------------------------------------------------------
// 2/5. C[M,N] = A[M,K](bf16) @ B[N,K]^T(bf16) + bias[N], fp32 out.
//      One 16x16 tile per wave; 8 waves/block along N; grid.y along M.
// ---------------------------------------------------------------------------
__global__ __launch_bounds__(256) void gemm_bf16_wmma(
    const unsigned short* __restrict__ A,
    const unsigned short* __restrict__ B,
    const float* __restrict__ bias,
    float* __restrict__ C,
    int N, int K)
{
  const int tid  = threadIdx.x;
  const int wave = tid >> 5, lane = tid & 31;
  const int n0   = (blockIdx.x * 8 + wave) * 16;
  const int m0   = blockIdx.y * 16;
  const int half = lane >> 4;          // 0: lanes 0-15, 1: lanes 16-31
  const int l16  = lane & 15;

  const unsigned short* arow = A + (size_t)(m0 + l16) * K;  // A row (M = l16)
  const unsigned short* brow = B + (size_t)(n0 + l16) * K;  // B col (N = l16)
  const int aklo = half * 8;           // A: K {0..7,16..23} / {8..15,24..31}
  const int bkhi = half * 8;           // B: K {0..7,16..23} / {8..15,24..31}

  v8f acc = {};
  for (int k = 0; k < K; k += 32) {
    __builtin_prefetch(arow + k + 512, 0, 1);
    __builtin_prefetch(brow + k + 512, 0, 1);
    U4 alo = *reinterpret_cast<const U4*>(arow + k + aklo);
    U4 ahi = *reinterpret_cast<const U4*>(arow + k + aklo + 16);
    U4 blo = *reinterpret_cast<const U4*>(brow + k + bkhi);
    U4 bhi = *reinterpret_cast<const U4*>(brow + k + bkhi + 16);
    v16bf af = make_frag(alo, ahi);
    v16bf bf = make_frag(blo, bhi);
    acc = __builtin_amdgcn_wmma_f32_16x16x32_bf16(
        false, af, false, bf, (short)0, acc, false, false);
  }
  const int   n    = n0 + l16;
  const float bn   = bias[n];
  const int   mrow = m0 + half * 8;    // C/D: lanes>=16 hold M = 8+r
#pragma unroll
  for (int r = 0; r < 8; ++r)
    C[(size_t)(mrow + r) * N + n] = acc[r] + bn;
}

// ---------------------------------------------------------------------------
// 3. Decoder input-gate precompute: two matvecs [3H,H]@[H] (tokens 0, tgt[1])
// ---------------------------------------------------------------------------
__global__ __launch_bounds__(256) void gi_dec_kernel(
    const float* __restrict__ w, const float* __restrict__ b,
    const float* __restrict__ emb, const int* __restrict__ targets,
    float* __restrict__ gi0, float* __restrict__ gic)
{
  const int wave = threadIdx.x >> 5, lane = threadIdx.x & 31;
  const int wid  = blockIdx.x * 8 + wave;          // 0 .. 2*H3-1
  const int row  = wid % H3;
  const int which = wid / H3;
  const int token = which ? targets[1] : 0;
  const float* x  = emb + (size_t)token * Hh;
  const float* wr = w + (size_t)row * Hh;
  float acc = 0.f;
  for (int k = lane; k < Hh; k += 32) acc += wr[k] * x[k];
#pragma unroll
  for (int m = 16; m; m >>= 1) acc += __shfl_xor(acc, m, 32);
  if (lane == 0) (which ? gic : gi0)[row] = acc + b[row];
}

// ---------------------------------------------------------------------------
// 4. Persistent GRU recurrence: 2048 encoder + 2048 decoder steps.
//    Grid-wide barrier per step via per-step counters (zeroed by memset).
// ---------------------------------------------------------------------------
static __device__ inline float dot8(U4 w, const float4& ha, const float4& hb) {
  float s;
  s  = bflo(w.a) * ha.x + bfhi(w.a) * ha.y;
  s += bflo(w.b) * ha.z + bfhi(w.b) * ha.w;
  s += bflo(w.c) * hb.x + bfhi(w.c) * hb.y;
  s += bflo(w.d) * hb.z + bfhi(w.d) * hb.w;
  return s;
}

__global__ __launch_bounds__(256) void gru_recurrence(
    const float* __restrict__ gi_enc,            // [S,3H]
    const float* __restrict__ gi_dec0,           // [3H]
    const float* __restrict__ gi_decc,           // [3H]
    const unsigned short* __restrict__ whh_enc,  // bf16 [3H,H]
    const unsigned short* __restrict__ whh_dec,  // bf16 [3H,H]
    const float* __restrict__ bhh_enc,           // [3H]
    const float* __restrict__ bhh_dec,           // [3H]
    float* __restrict__ hbuf,                    // [2][H] (zeroed)
    unsigned short* __restrict__ dec_out,        // bf16 [T,H]
    unsigned int* __restrict__ ctr)              // [4096] (zeroed)
{
  __shared__ float hs[Hh];
  const int tid  = threadIdx.x;
  const int wave = tid >> 5, lane = tid & 31;
  const int wid  = blockIdx.x * 8 + wave;        // 0..511 -> 4 j's each

  int parity = 0;
  for (int step = 0; step < SS + TT; ++step) {
    const bool enc = step < SS;
    // ---- stage h_{t-1} into LDS (8 KB) via async DMA (ASYNCcnt path) ----
    const float* hin = hbuf + parity * Hh;
#if HAVE_ASYNC_LDS
    {
      const float* g0 = hin + tid * 4;           // 16 B per lane, two halves
      __builtin_amdgcn_global_load_async_to_lds_b128(
          AS1V4(g0), AS3V4(&hs[tid * 4]), 0, 0);
      __builtin_amdgcn_global_load_async_to_lds_b128(
          AS1V4(g0 + Hh / 2), AS3V4(&hs[tid * 4 + Hh / 2]), 0, 0);
#if __has_builtin(__builtin_amdgcn_s_wait_asynccnt)
      __builtin_amdgcn_s_wait_asynccnt(0);
#else
      asm volatile("s_wait_asynccnt 0x0" ::: "memory");
#endif
    }
#else
#pragma unroll
    for (int i = 0; i < Hh / 256; ++i) hs[tid + i * 256] = hin[tid + i * 256];
#endif
    __syncthreads();

    const unsigned short* whh = enc ? whh_enc : whh_dec;
    const float* bhh = enc ? bhh_enc : bhh_dec;
    const float* gi  = enc ? (gi_enc + (size_t)step * H3)
                           : ((step == SS) ? gi_dec0 : gi_decc);
    float* hout = hbuf + (parity ^ 1) * Hh;

#pragma unroll
    for (int jj = 0; jj < 4; ++jj) {
      const int j = wid * 4 + jj;
      const unsigned short* wr = whh + (size_t)j * Hh;
      const unsigned short* wz = whh + (size_t)(j + Hh) * Hh;
      const unsigned short* wn = whh + (size_t)(j + 2 * Hh) * Hh;
      float ar = 0.f, az = 0.f, an = 0.f;
#pragma unroll
      for (int it = 0; it < Hh / 256; ++it) {
        const int k = it * 256 + lane * 8;
        const float4 ha = *reinterpret_cast<const float4*>(&hs[k]);
        const float4 hb = *reinterpret_cast<const float4*>(&hs[k + 4]);
        ar += dot8(*reinterpret_cast<const U4*>(wr + k), ha, hb);
        az += dot8(*reinterpret_cast<const U4*>(wz + k), ha, hb);
        an += dot8(*reinterpret_cast<const U4*>(wn + k), ha, hb);
      }
#pragma unroll
      for (int m = 16; m; m >>= 1) {
        ar += __shfl_xor(ar, m, 32);
        az += __shfl_xor(az, m, 32);
        an += __shfl_xor(an, m, 32);
      }
      if (lane == 0) {
        const float ghr = ar + bhh[j];
        const float ghz = az + bhh[j + Hh];
        const float ghn = an + bhh[j + 2 * Hh];
        const float r = 1.f / (1.f + expf(-(gi[j] + ghr)));
        const float z = 1.f / (1.f + expf(-(gi[j + Hh] + ghz)));
        const float n = tanhf(gi[j + 2 * Hh] + r * ghn);
        const float hn = (1.f - z) * n + z * hs[j];
        hout[j] = hn;
        if (!enc) dec_out[(size_t)(step - SS) * Hh + j] = f2bf(hn);
      }
    }

    // ---- grid-wide barrier: per-step counter, no sense reversal needed ----
    __threadfence();           // publish hout (and dec_out) to device scope
    __syncthreads();
    if (tid == 0) {
      __hip_atomic_fetch_add(&ctr[step], 1u, __ATOMIC_RELEASE,
                             __HIP_MEMORY_SCOPE_AGENT);
      while (__hip_atomic_load(&ctr[step], __ATOMIC_ACQUIRE,
                               __HIP_MEMORY_SCOPE_AGENT) < (unsigned)NBLK)
        __builtin_amdgcn_s_sleep(2);
    }
    __syncthreads();
    __threadfence();           // acquire: refresh view of hbuf
    parity ^= 1;
  }
}

// ---------------------------------------------------------------------------
// Host launcher
// ---------------------------------------------------------------------------
extern "C" void kernel_launch(void* const* d_in, const int* in_sizes, int n_in,
                              void* d_out, int out_size, void* d_ws, size_t ws_size,
                              hipStream_t stream) {
  (void)in_sizes; (void)n_in; (void)out_size; (void)ws_size;
  const int*   inputs   = (const int*)  d_in[0];
  const int*   targets  = (const int*)  d_in[1];
  const float* emb      = (const float*)d_in[2];
  const float* enc_b_ih = (const float*)d_in[5];
  const float* enc_b_hh = (const float*)d_in[6];
  const float* dec_w_ih = (const float*)d_in[7];
  const float* dec_b_ih = (const float*)d_in[9];
  const float* dec_b_hh = (const float*)d_in[10];
  const float* fc_b     = (const float*)d_in[12];

  char* ws = (char*)d_ws;
  // workspace layout (bytes, 256-aligned by construction)
  const size_t OFF_XBF   = 0;                      //  8 MB  X bf16 [S,H]
  const size_t OFF_WIHBF = OFF_XBF   + 8388608;    // 24 MB  enc_w_ih bf16
  const size_t OFF_WHHE  = OFF_WIHBF + 25165824;   // 24 MB  enc_w_hh bf16
  const size_t OFF_WHHD  = OFF_WHHE  + 25165824;   // 24 MB  dec_w_hh bf16
  const size_t OFF_FCBF  = OFF_WHHD  + 25165824;   //  1 MB  fc_w bf16
  const size_t OFF_GIENC = OFF_FCBF  + 1048576;    // 48 MB  GI_enc fp32 [S,3H]
  const size_t OFF_GI0   = OFF_GIENC + 50331648;   // 24 KB
  const size_t OFF_GIC   = OFF_GI0   + 24576;      // 24 KB
  const size_t OFF_CTR   = OFF_GIC   + 24576;      // 16 KB  barrier counters
  const size_t OFF_HBUF  = OFF_CTR   + 16384;      // 16 KB  h ping-pong
  const size_t OFF_DECBF = OFF_HBUF  + 16384;      //  8 MB  dec_hs bf16 [T,H]

  unsigned short* xbf   = (unsigned short*)(ws + OFF_XBF);
  unsigned short* wihbf = (unsigned short*)(ws + OFF_WIHBF);
  unsigned short* whhe  = (unsigned short*)(ws + OFF_WHHE);
  unsigned short* whhd  = (unsigned short*)(ws + OFF_WHHD);
  unsigned short* fcbf  = (unsigned short*)(ws + OFF_FCBF);
  float*          gienc = (float*)(ws + OFF_GIENC);
  float*          gi0   = (float*)(ws + OFF_GI0);
  float*          gic   = (float*)(ws + OFF_GIC);
  unsigned int*   ctr   = (unsigned int*)(ws + OFF_CTR);
  float*          hbuf  = (float*)(ws + OFF_HBUF);
  unsigned short* decbf = (unsigned short*)(ws + OFF_DECBF);

  // zero barrier counters + h0 (contiguous 32 KB)
  (void)hipMemsetAsync(ws + OFF_CTR, 0, 32768, stream);

  // 1. gather/convert (42,467,328 elements)
  prep_convert<<<dim3(165888), 256, 0, stream>>>(
      inputs, emb,
      (const float*)d_in[3], (const float*)d_in[4],
      (const float*)d_in[8], (const float*)d_in[11],
      xbf, wihbf, whhe, whhd, fcbf);

  // 2. GI_enc[S,3H] = X @ enc_w_ih^T + b_ih   (M=2048, N=6144, K=2048)
  gemm_bf16_wmma<<<dim3(H3 / 128, SS / 16), 256, 0, stream>>>(
      xbf, wihbf, enc_b_ih, gienc, H3, Hh);

  // 3. decoder input gates (2 matvecs)
  gi_dec_kernel<<<dim3(2 * H3 / 8), 256, 0, stream>>>(
      dec_w_ih, dec_b_ih, emb, targets, gi0, gic);

  // 4. sequential recurrence (persistent, grid barrier per step)
  gru_recurrence<<<dim3(NBLK), 256, 0, stream>>>(
      gienc, gi0, gic, whhe, whhd, enc_b_hh, dec_b_hh, hbuf, decbf, ctr);

  // 5. out[T,V] = dec_hs @ fc_w^T + fc_b   (M=2048, N=256, K=2048)
  gemm_bf16_wmma<<<dim3(VV / 128, TT / 16), 256, 0, stream>>>(
      decbf, fcbf, fc_b, (float*)d_out, VV, Hh);
}
